// StreamingISTFT_49460843381095
// MI455X (gfx1250) — matrix operand order repaired and checked by reading, not compiled, not run on hardware
//
#include <hip/hip_runtime.h>
#include <hip/hip_bf16.h>

typedef _Float16 v16h __attribute__((ext_vector_type(16)));
typedef _Float16 v8h  __attribute__((ext_vector_type(8)));
typedef float    v8f  __attribute__((ext_vector_type(8)));

#define NFFT      2048
#define NFREQ     1024
#define NT        2048
#define NB        8
#define HOP       512
#define OUT_PER_B (HOP * NT)          // 1048576

// Fragment-linear layouts (per 16x32 A-frag / 32x16 B-frag, 512 halves each):
//   A: h[j] of lane = Basis[n][k], n = r*16 + (lane&15),
//      k = c*32 + (lane>=16)*16 + j
//   B: h[j] of lane = Spec[k][t],  t = tb*16 + (lane&15),
//      k = c*32 + (lane>=16)*16 + j
// so each lane's 16 fragment halves are 32 contiguous bytes at frag*1024 + lane*32.

// --- Kernel 1: DFT basis (cos | -sin), f16, stored fragment-linear ---
__global__ __launch_bounds__(256) void istft_k_basis(_Float16* __restrict__ basis) {
    const float OMEGA = 0.00306796157577128f;      // 2*pi/2048
    unsigned e = blockIdx.x * 256 + threadIdx.x;   // over 2048*2048
    unsigned j    = e & 15u;
    unsigned lane = (e >> 4) & 31u;
    unsigned c    = (e >> 9) & 63u;
    unsigned r    = e >> 15;                       // 0..127
    unsigned n = r * 16 + (lane & 15u);            // sample index l
    unsigned k = c * 32 + ((lane >> 4) << 4) + j;
    float val;
    if (k < NFREQ) {
        unsigned ph = (n * k) & 2047u;             // exact mod-2048 reduction
        float sc = (k == 0) ? (1.0f / 2048.0f) : (2.0f / 2048.0f);
        val = sc * __cosf((float)ph * OMEGA);
    } else {
        unsigned kk = k - NFREQ;
        if (kk == 0) {
            val = 0.0f;                            // Im(DC) ignored by irfft (c2r)
        } else {
            unsigned ph = (n * kk) & 2047u;
            val = -(2.0f / 2048.0f) * __sinf((float)ph * OMEGA);
        }
    }
    basis[e] = (_Float16)val;
}

// --- Kernel 2: denormalize + transpose -> f16 spec, fragment-linear per batch ---
// tile: 64 f x 64 t, LDS transpose, coalesced read (t-contig) and write (k-contig).
__global__ __launch_bounds__(256) void istft_k_spec(const float* __restrict__ in,
                                                    _Float16* __restrict__ spec) {
    __shared__ float lre[64][65];
    __shared__ float lim[64][65];
    const int b  = blockIdx.z;
    const int f0 = blockIdx.y * 64;
    const int t0 = blockIdx.x * 64;
    const float* pre = in + (((size_t)b * 2 + 0) * NFREQ + f0) * NT + t0;
    const float* pim = in + (((size_t)b * 2 + 1) * NFREQ + f0) * NT + t0;
    for (int i = threadIdx.x; i < 64 * 64; i += 256) {
        int fl = i >> 6, tl = i & 63;
        lre[fl][tl] = pre[(size_t)fl * NT + tl];
        lim[fl][tl] = pim[(size_t)fl * NT + tl];
    }
    __syncthreads();
    _Float16* specB = spec + (size_t)b * NFFT * NT;
    for (int i = threadIdx.x; i < 64 * 64; i += 256) {
        int tl = i >> 6, fl = i & 63;
        float re = lre[fl][tl], im = lim[fl][tl];
        // spec = (re^2+im^2)^P * beta^(-2P-1) * (re, im);  P = (1/0.65 - 1)/2
        float s = __powf(re * re + im * im, 0.26923076923076916f) * 75.81706f;
        int t = t0 + tl;
        int f = f0 + fl;
        // fragment-linear address for (k, t)
        #pragma unroll
        for (int p = 0; p < 2; ++p) {
            int k  = f + p * NFREQ;               // Re plane then Im plane
            int r5 = k & 31;
            int ln = (t & 15) | ((r5 >> 4) << 4);
            size_t o = (((size_t)(t >> 4) * 64 + (k >> 5)) << 9) + (ln << 4) + (r5 & 15);
            specB[o] = (_Float16)(s * (p ? im : re));
        }
    }
}

// --- Kernel 3: WMMA GEMM (frames = Basis x Spec), no LDS, fused window + OLA ---
// grid (8,16,8), block 256 (8 waves: 2 wm x 4 wn). Block tile 128(l) x 256(t).
// Each wave: 4(M) x 4(N) fragments -> 16 wmma per 32-wide K step against
// 16 b128 loads (A reused 4x, B reused 4x): ~32 B / matrix-cycle per wave.
__global__ __launch_bounds__(256) void istft_k_gemm(const _Float16* __restrict__ basis,
                                                    const _Float16* __restrict__ spec,
                                                    const float* __restrict__ invw,
                                                    float* __restrict__ out) {
    const int tid  = threadIdx.x;
    const int lane = tid & 31;
    const int wave = tid >> 5;
    const int wm   = wave & 1;             // 0..1 : 64-row strip
    const int wn   = wave >> 1;            // 0..3 : 64-col strip
    const int bn0  = blockIdx.x * 256;     // t tile base
    const int bm0  = blockIdx.y * 128;     // l tile base
    const int b    = blockIdx.z;
    const int l15  = lane & 15;
    const int hi   = lane >> 4;

    // per-lane fragment stream bases (halves); each K step advances 512 halves
    const _Float16* ap[4];
    #pragma unroll
    for (int mi = 0; mi < 4; ++mi) {
        int r = (bm0 >> 4) + wm * 4 + mi;
        ap[mi] = basis + (((size_t)r * 64) << 9) + (lane << 4);
    }
    const _Float16* bp[4];
    #pragma unroll
    for (int ni = 0; ni < 4; ++ni) {
        int tb = (bn0 >> 4) + wn * 4 + ni;
        bp[ni] = spec + (((size_t)b * 128 + tb) * 64 << 9) + (lane << 4);
    }

    v8f acc[4][4] = {};

    #pragma unroll 1
    for (int kb = 0; kb < NFFT; kb += 32) {
        const size_t ko = (size_t)kb << 4;     // kb/32 * 512 halves
        v16h afr[4], bfr[4];
        #pragma unroll
        for (int mi = 0; mi < 4; ++mi) {
            v8h lo = *(const v8h*)(ap[mi] + ko);
            v8h hv = *(const v8h*)(ap[mi] + ko + 8);
            #pragma unroll
            for (int j = 0; j < 8; ++j) { afr[mi][j] = lo[j]; afr[mi][j + 8] = hv[j]; }
        }
        #pragma unroll
        for (int ni = 0; ni < 4; ++ni) {
            v8h lo = *(const v8h*)(bp[ni] + ko);
            v8h hv = *(const v8h*)(bp[ni] + ko + 8);
            #pragma unroll
            for (int j = 0; j < 8; ++j) { bfr[ni][j] = lo[j]; bfr[ni][j + 8] = hv[j]; }
        }
        if (kb < NFFT - 128) {                 // pull next tiles toward the WGP
            const size_t po = ko + (128 << 4);
            #pragma unroll
            for (int mi = 0; mi < 4; ++mi) __builtin_prefetch(ap[mi] + po, 0, 3);
            #pragma unroll
            for (int ni = 0; ni < 4; ++ni) __builtin_prefetch(bp[ni] + po, 0, 3);
        }
        #pragma unroll
        for (int mi = 0; mi < 4; ++mi)
            #pragma unroll
            for (int ni = 0; ni < 4; ++ni)
                acc[mi][ni] = __builtin_amdgcn_wmma_f32_16x16x32_f16(
                    false, afr[mi], false, bfr[ni], (short)0, acc[mi][ni], false, false);
    }

    // Epilogue: window multiply + overlap-add (truncate at HOP*NT like the reference)
    float* outB = out + (size_t)b * OUT_PER_B;
    #pragma unroll
    for (int mi = 0; mi < 4; ++mi) {
        #pragma unroll
        for (int g = 0; g < 8; ++g) {
            int l = bm0 + wm * 64 + mi * 16 + g + hi * 8;
            float w = invw[l];
            #pragma unroll
            for (int ni = 0; ni < 4; ++ni) {
                int tcol = bn0 + wn * 64 + ni * 16 + l15;
                int gi = tcol * HOP + l;
                if (gi < OUT_PER_B) {
                    atomicAdd(outB + gi, acc[mi][ni][g] * w);
                }
            }
        }
    }
}

extern "C" void kernel_launch(void* const* d_in, const int* in_sizes, int n_in,
                              void* d_out, int out_size, void* d_ws, size_t ws_size,
                              hipStream_t stream) {
    const float* stft = (const float*)d_in[0];   // [8,2,1024,2048] f32
    const float* invw = (const float*)d_in[1];   // [2048] f32
    float* out = (float*)d_out;                  // [8, 1048576] f32

    _Float16* basis = (_Float16*)d_ws;                                    // 8 MB
    _Float16* spec  = (_Float16*)((char*)d_ws + (size_t)NFFT * NFFT * 2); // 64 MB

    hipMemsetAsync(d_out, 0, (size_t)out_size * sizeof(float), stream);

    istft_k_basis<<<(NFFT * NFFT) / 256, 256, 0, stream>>>(basis);

    dim3 gs(NT / 64, NFREQ / 64, NB);
    istft_k_spec<<<gs, 256, 0, stream>>>(stft, spec);

    dim3 gg(NT / 256, NFFT / 128, NB);
    istft_k_gemm<<<gg, 256, 0, stream>>>(basis, spec, invw, out);
}
